// SpatialAwareTrans_42502996361537
// MI455X (gfx1250) — matrix operand order repaired and checked by reading, not compiled, not run on hardware
//
#include <hip/hip_runtime.h>
#include <stdint.h>

typedef __attribute__((ext_vector_type(16))) __bf16 v16bf;
typedef __attribute__((ext_vector_type(8)))  __bf16 v8bf;
typedef __attribute__((ext_vector_type(4)))  __bf16 v4bf;
typedef __attribute__((ext_vector_type(8)))  float  v8f;

#define DIMC 128
#define NWIN 1024
#define NTOK 85
#define TTOT 87040   // 1024 windows * 85 tokens

__device__ __forceinline__ __bf16 f2bf(float f) { return (__bf16)f; }

union Frag { v16bf v; v8bf h[2]; uint32_t u[8]; };

__device__ __forceinline__ v8f wmma_bf16(v16bf a, v16bf b, v8f c) {
  return __builtin_amdgcn_wmma_f32_16x16x32_bf16(false, a, false, b, (short)0, c,
                                                 false, false);
}

// A-fragment (16x32 bf16, MxK): lane%16 = row, elems 0-7 at K=k0+(lane>=16?8:0),
// elems 8-15 at that +16.  (CDNA5 ISA 7.12.2)
__device__ __forceinline__ v16bf load_a_bf16(const __bf16* A, int ld, int m0, int k0, int lane) {
  const __bf16* p = A + (size_t)(m0 + (lane & 15)) * ld + k0 + ((lane >> 4) << 3);
  Frag f;
  f.h[0] = *(const v8bf*)p;
  f.h[1] = *(const v8bf*)(p + 16);
  return f.v;
}

// Same fragment but A source is fp32 (hardware v_cvt_pk_bf16_f32 on load).
__device__ __forceinline__ v16bf load_a_f32(const float* rowp, int k0, int lane) {
  const float* p = rowp + k0 + ((lane >> 4) << 3);
  Frag f;
#pragma unroll
  for (int e = 0; e < 8; ++e) { f.v[e] = f2bf(p[e]); f.v[e + 8] = f2bf(p[e + 16]); }
  return f.v;
}

// B-fragment (32x16 bf16, KxN) from pre-transposed weights Wt[N][K]:
// lane%16 = column, elems 0-7 at K=k0+(lane>=16?8:0), elems 8-15 at that +16.
__device__ __forceinline__ v16bf load_b_bf16(const __bf16* Wt, int ld, int n0, int k0, int lane) {
  const __bf16* p = Wt + (size_t)(n0 + (lane & 15)) * ld + k0 + ((lane >> 4) << 3);
  Frag f;
  f.h[0] = *(const v8bf*)p;
  f.h[1] = *(const v8bf*)(p + 16);
  return f.v;
}

// window-partition token index: row of (B,HS,HS) -> position in [1024][85] token buf
__device__ __forceinline__ int token_of(int row, int lhs, int lws, int off) {
  int b = row >> (2 * lhs);
  int rem = row & ((1 << (2 * lhs)) - 1);
  int h = rem >> lhs;
  int w = rem & ((1 << lhs) - 1);
  int win = ((b << 3) + (h >> lws)) * 8 + (w >> lws);
  int ws1 = (1 << lws) - 1;
  int pos = off + ((h & ws1) << lws) + (w & ws1);
  return win * NTOK + pos;
}

#define ACC_ZERO {0.f,0.f,0.f,0.f,0.f,0.f,0.f,0.f}

// ---------------- weight convert + transpose: Wt[n][k] = bf16(W[k][n]) -------
__global__ void wtrans_kernel(const float* __restrict__ W, __bf16* __restrict__ Wt,
                              int K, int N) {
  int idx = blockIdx.x * 256 + threadIdx.x;
  if (idx >= K * N) return;
  int k = idx / N, n = idx - k * N;
  Wt[(size_t)n * K + k] = f2bf(W[idx]);
}

// ---------------- per-scale fc GEMM + window-partition scatter ---------------
// one wave computes a 32x64 strip: 2 A-frags x 4 B-frags -> 8 WMMAs per k-step
__global__ void fc_gemm(const float* __restrict__ X, const __bf16* __restrict__ Wt,
                        const float* __restrict__ bias, float* __restrict__ xcat,
                        int CH, int lhs, int lws, int off) {
  int lane = threadIdx.x & 31;
  int m0 = (blockIdx.x * 4 + (threadIdx.x >> 5)) << 5;
  int n0 = blockIdx.y << 6;
  const float* rowp0 = X + (size_t)(m0 + (lane & 15)) * CH;
  const float* rowp1 = rowp0 + (size_t)16 * CH;
  v8f acc[2][4] = {{ACC_ZERO, ACC_ZERO, ACC_ZERO, ACC_ZERO},
                   {ACC_ZERO, ACC_ZERO, ACC_ZERO, ACC_ZERO}};
  for (int k0 = 0; k0 < CH; k0 += 32) {
    v16bf a0 = load_a_f32(rowp0, k0, lane);
    v16bf a1 = load_a_f32(rowp1, k0, lane);
    v16bf b0 = load_b_bf16(Wt, CH, n0,      k0, lane);
    v16bf b1 = load_b_bf16(Wt, CH, n0 + 16, k0, lane);
    v16bf b2 = load_b_bf16(Wt, CH, n0 + 32, k0, lane);
    v16bf b3 = load_b_bf16(Wt, CH, n0 + 48, k0, lane);
    acc[0][0] = wmma_bf16(a0, b0, acc[0][0]);
    acc[0][1] = wmma_bf16(a0, b1, acc[0][1]);
    acc[0][2] = wmma_bf16(a0, b2, acc[0][2]);
    acc[0][3] = wmma_bf16(a0, b3, acc[0][3]);
    acc[1][0] = wmma_bf16(a1, b0, acc[1][0]);
    acc[1][1] = wmma_bf16(a1, b1, acc[1][1]);
    acc[1][2] = wmma_bf16(a1, b2, acc[1][2]);
    acc[1][3] = wmma_bf16(a1, b3, acc[1][3]);
  }
#pragma unroll
  for (int mi = 0; mi < 2; ++mi)
#pragma unroll
    for (int t = 0; t < 4; ++t) {
      int col = n0 + (t << 4) + (lane & 15);
      float bv = bias[col];
#pragma unroll
      for (int i = 0; i < 8; ++i) {
        int row = m0 + (mi << 4) + i + ((lane >> 4) << 3);
        xcat[(size_t)token_of(row, lhs, lws, off) * DIMC + col] = acc[mi][t][i] + bv;
      }
    }
}

// ---------------- LayerNorm over 128 channels, one wave per token ------------
__global__ void ln_kernel(const float* __restrict__ x, const float* __restrict__ g,
                          const float* __restrict__ b, __bf16* __restrict__ out) {
  int t = blockIdx.x * 8 + (threadIdx.x >> 5);
  int lane = threadIdx.x & 31;
  const float4 v = *(const float4*)(x + (size_t)t * DIMC + lane * 4);
  float s = v.x + v.y + v.z + v.w;
  s += __shfl_xor(s, 1, 32);  s += __shfl_xor(s, 2, 32);
  s += __shfl_xor(s, 4, 32);  s += __shfl_xor(s, 8, 32);
  s += __shfl_xor(s, 16, 32);
  float mean = s * (1.f / 128.f);
  float dx = v.x - mean, dy = v.y - mean, dz = v.z - mean, dw = v.w - mean;
  float q = dx * dx + dy * dy + dz * dz + dw * dw;
  q += __shfl_xor(q, 1, 32);  q += __shfl_xor(q, 2, 32);
  q += __shfl_xor(q, 4, 32);  q += __shfl_xor(q, 8, 32);
  q += __shfl_xor(q, 16, 32);
  float r = rsqrtf(q * (1.f / 128.f) + 1e-6f);
  const float4 gv = *(const float4*)(g + lane * 4);
  const float4 bv = *(const float4*)(b + lane * 4);
  v4bf o;
  o[0] = f2bf(dx * r * gv.x + bv.x);
  o[1] = f2bf(dy * r * gv.y + bv.y);
  o[2] = f2bf(dz * r * gv.z + bv.z);
  o[3] = f2bf(dw * r * gv.w + bv.w);
  *(v4bf*)(out + (size_t)t * DIMC + lane * 4) = o;
}

// ---------------- generic bf16-out GEMM, 32x64 strip per wave ----------------
__global__ void gemm_bf16(const __bf16* __restrict__ A, const __bf16* __restrict__ Wt,
                          const float* __restrict__ bias, __bf16* __restrict__ out,
                          int N, int K, int relu) {
  int lane = threadIdx.x & 31;
  int m0 = (blockIdx.x * 4 + (threadIdx.x >> 5)) << 5;
  int n0 = blockIdx.y << 6;
  v8f acc[2][4] = {{ACC_ZERO, ACC_ZERO, ACC_ZERO, ACC_ZERO},
                   {ACC_ZERO, ACC_ZERO, ACC_ZERO, ACC_ZERO}};
  for (int k0 = 0; k0 < K; k0 += 32) {
    v16bf a0 = load_a_bf16(A, K, m0,      k0, lane);
    v16bf a1 = load_a_bf16(A, K, m0 + 16, k0, lane);
    v16bf b0 = load_b_bf16(Wt, K, n0,      k0, lane);
    v16bf b1 = load_b_bf16(Wt, K, n0 + 16, k0, lane);
    v16bf b2 = load_b_bf16(Wt, K, n0 + 32, k0, lane);
    v16bf b3 = load_b_bf16(Wt, K, n0 + 48, k0, lane);
    acc[0][0] = wmma_bf16(a0, b0, acc[0][0]);
    acc[0][1] = wmma_bf16(a0, b1, acc[0][1]);
    acc[0][2] = wmma_bf16(a0, b2, acc[0][2]);
    acc[0][3] = wmma_bf16(a0, b3, acc[0][3]);
    acc[1][0] = wmma_bf16(a1, b0, acc[1][0]);
    acc[1][1] = wmma_bf16(a1, b1, acc[1][1]);
    acc[1][2] = wmma_bf16(a1, b2, acc[1][2]);
    acc[1][3] = wmma_bf16(a1, b3, acc[1][3]);
  }
#pragma unroll
  for (int mi = 0; mi < 2; ++mi)
#pragma unroll
    for (int t = 0; t < 4; ++t) {
      int col = n0 + (t << 4) + (lane & 15);
      float bv = bias ? bias[col] : 0.f;
#pragma unroll
      for (int i = 0; i < 8; ++i) {
        int row = m0 + (mi << 4) + i + ((lane >> 4) << 3);
        float vv = acc[mi][t][i] + bv;
        if (relu) vv = fmaxf(vv, 0.f);
        out[(size_t)row * N + col] = f2bf(vv);
      }
    }
}

// ---------------- fp32-out GEMM with in-place residual, 32x64 strip ----------
__global__ void gemm_f32_resid(const __bf16* __restrict__ A, const __bf16* __restrict__ Wt,
                               const float* __restrict__ bias, float* __restrict__ out,
                               int N, int K) {
  int lane = threadIdx.x & 31;
  int m0 = (blockIdx.x * 4 + (threadIdx.x >> 5)) << 5;
  int n0 = blockIdx.y << 6;
  v8f acc[2][4] = {{ACC_ZERO, ACC_ZERO, ACC_ZERO, ACC_ZERO},
                   {ACC_ZERO, ACC_ZERO, ACC_ZERO, ACC_ZERO}};
  for (int k0 = 0; k0 < K; k0 += 32) {
    v16bf a0 = load_a_bf16(A, K, m0,      k0, lane);
    v16bf a1 = load_a_bf16(A, K, m0 + 16, k0, lane);
    v16bf b0 = load_b_bf16(Wt, K, n0,      k0, lane);
    v16bf b1 = load_b_bf16(Wt, K, n0 + 16, k0, lane);
    v16bf b2 = load_b_bf16(Wt, K, n0 + 32, k0, lane);
    v16bf b3 = load_b_bf16(Wt, K, n0 + 48, k0, lane);
    acc[0][0] = wmma_bf16(a0, b0, acc[0][0]);
    acc[0][1] = wmma_bf16(a0, b1, acc[0][1]);
    acc[0][2] = wmma_bf16(a0, b2, acc[0][2]);
    acc[0][3] = wmma_bf16(a0, b3, acc[0][3]);
    acc[1][0] = wmma_bf16(a1, b0, acc[1][0]);
    acc[1][1] = wmma_bf16(a1, b1, acc[1][1]);
    acc[1][2] = wmma_bf16(a1, b2, acc[1][2]);
    acc[1][3] = wmma_bf16(a1, b3, acc[1][3]);
  }
#pragma unroll
  for (int mi = 0; mi < 2; ++mi)
#pragma unroll
    for (int t = 0; t < 4; ++t) {
      int col = n0 + (t << 4) + (lane & 15);
      float bv = bias ? bias[col] : 0.f;
#pragma unroll
      for (int i = 0; i < 8; ++i) {
        int row = m0 + (mi << 4) + i + ((lane >> 4) << 3);
        size_t idx = (size_t)row * N + col;
        out[idx] = out[idx] + acc[mi][t][i] + bv;
      }
    }
}

// ---------------- windowed attention: 1 block/window, 1 wave/head ------------
// qkv: [TTOT][384] bf16 (q|k|v, each [8 heads][16]);  aout: [TTOT][128] bf16
__global__ void attn_kernel(const __bf16* __restrict__ qkv, __bf16* __restrict__ aout) {
  __shared__ __align__(32) __bf16 vT[8][16][96];   // V transposed: [head][d][m]
  __shared__ __align__(32) __bf16 Pt[8][16][96];   // softmax(P) row-tile
  int hh = threadIdx.x >> 5;
  int lane = threadIdx.x & 31;
  int lh = lane >> 4;        // half-wave select
  int ln = lane & 15;
  size_t base = (size_t)blockIdx.x * NTOK;
  const __bf16* qbase = qkv + base * 384 + hh * 16;
  const __bf16* kbase = qbase + 128;
  const __bf16* vbase = qbase + 256;

  // stage V transposed into LDS (zero-pad m>=85 so K-padding multiplies by 0)
  for (int idx = lane; idx < 16 * 96; idx += 32) {
    int d = idx & 15, m = idx >> 4;
    __bf16 val = __builtin_bit_cast(__bf16, (uint16_t)0);
    if (m < NTOK) val = vbase[(size_t)m * 384 + d];
    vT[hh][d][m] = val;
  }
  Frag zero;
#pragma unroll
  for (int e = 0; e < 8; ++e) zero.u[e] = 0;

  for (int I = 0; I < 6; ++I) {               // 6 row-tiles of 16 (96 >= 85)
    int qn = I * 16 + ln; if (qn > NTOK - 1) qn = NTOK - 1;
    Frag qa;                                   // head_dim 16 padded to K=32
    qa.h[0] = *(const v8bf*)(qbase + (size_t)qn * 384 + (lh << 3));
    qa.h[1] = zero.h[0];
    v8f att[6];
#pragma unroll
    for (int J = 0; J < 6; ++J) {
      int km = J * 16 + ln; if (km > NTOK - 1) km = NTOK - 1;
      Frag kf;                                 // elems 8-15 carry K=16..31 -> zero
      kf.h[0] = *(const v8bf*)(kbase + (size_t)km * 384 + (lh << 3));
      kf.h[1] = zero.h[0];
      v16bf kb = kf.v;
      v8f z = ACC_ZERO;
      att[J] = wmma_bf16(qa.v, kb, z);
    }
    // mask out-of-range key columns
#pragma unroll
    for (int J = 0; J < 6; ++J)
      if (J * 16 + ln > NTOK - 1) {
#pragma unroll
        for (int i = 0; i < 8; ++i) att[J][i] = -1e30f;
      }
    // row softmax: rows live on 16-lane halves; xor 1/2/4/8 stays in-half
    float rs[8];
#pragma unroll
    for (int i = 0; i < 8; ++i) {
      float mx = att[0][i];
#pragma unroll
      for (int J = 1; J < 6; ++J) mx = fmaxf(mx, att[J][i]);
      mx = fmaxf(mx, __shfl_xor(mx, 1, 32));
      mx = fmaxf(mx, __shfl_xor(mx, 2, 32));
      mx = fmaxf(mx, __shfl_xor(mx, 4, 32));
      mx = fmaxf(mx, __shfl_xor(mx, 8, 32));
      float s = 0.f;
#pragma unroll
      for (int J = 0; J < 6; ++J) {
        float p = __expf(att[J][i] - mx);
        att[J][i] = p;
        s += p;
      }
      s += __shfl_xor(s, 1, 32);
      s += __shfl_xor(s, 2, 32);
      s += __shfl_xor(s, 4, 32);
      s += __shfl_xor(s, 8, 32);
      rs[i] = s;
    }
    // spill P tile to LDS so it can be re-read in A-fragment layout
#pragma unroll
    for (int i = 0; i < 8; ++i) {
      int r = i + (lh << 3);
#pragma unroll
      for (int J = 0; J < 6; ++J)
        Pt[hh][r][J * 16 + ln] = f2bf(att[J][i]);
    }
    // out = P @ V  (K = 96 -> 3 wmma steps of 32)
    v8f oacc = ACC_ZERO;
#pragma unroll
    for (int Jk = 0; Jk < 3; ++Jk) {
      Frag pa;
      const __bf16* pp = &Pt[hh][ln][Jk * 32 + (lh << 3)];
      pa.h[0] = *(const v8bf*)pp;
      pa.h[1] = *(const v8bf*)(pp + 16);
      Frag vf;
      const __bf16* vp = &vT[hh][ln][Jk * 32 + (lh << 3)];
      vf.h[0] = *(const v8bf*)vp;
      vf.h[1] = *(const v8bf*)(vp + 16);
      v16bf vb = vf.v;
      oacc = wmma_bf16(pa.v, vb, oacc);
    }
#pragma unroll
    for (int i = 0; i < 8; ++i) {
      int r = I * 16 + i + (lh << 3);
      if (r < NTOK)
        aout[(base + r) * DIMC + hh * 16 + ln] = f2bf(oacc[i] / rs[i]);
    }
  }
}

// ---------------- per-scale reverse GEMM + window-reverse gather -------------
__global__ void rev_gemm(const float* __restrict__ xf, const __bf16* __restrict__ Wt,
                         const float* __restrict__ bias, float* __restrict__ out,
                         int CHN, int lhs, int lws, int off) {
  int lane = threadIdx.x & 31;
  int m0 = (blockIdx.x * 4 + (threadIdx.x >> 5)) << 5;
  int n0 = blockIdx.y << 6;
  const float* rowp0 = xf + (size_t)token_of(m0 + (lane & 15), lhs, lws, off) * DIMC;
  const float* rowp1 = xf + (size_t)token_of(m0 + 16 + (lane & 15), lhs, lws, off) * DIMC;
  v8f acc[2][4] = {{ACC_ZERO, ACC_ZERO, ACC_ZERO, ACC_ZERO},
                   {ACC_ZERO, ACC_ZERO, ACC_ZERO, ACC_ZERO}};
  for (int k0 = 0; k0 < DIMC; k0 += 32) {
    v16bf a0 = load_a_f32(rowp0, k0, lane);
    v16bf a1 = load_a_f32(rowp1, k0, lane);
    v16bf b0 = load_b_bf16(Wt, DIMC, n0,      k0, lane);
    v16bf b1 = load_b_bf16(Wt, DIMC, n0 + 16, k0, lane);
    v16bf b2 = load_b_bf16(Wt, DIMC, n0 + 32, k0, lane);
    v16bf b3 = load_b_bf16(Wt, DIMC, n0 + 48, k0, lane);
    acc[0][0] = wmma_bf16(a0, b0, acc[0][0]);
    acc[0][1] = wmma_bf16(a0, b1, acc[0][1]);
    acc[0][2] = wmma_bf16(a0, b2, acc[0][2]);
    acc[0][3] = wmma_bf16(a0, b3, acc[0][3]);
    acc[1][0] = wmma_bf16(a1, b0, acc[1][0]);
    acc[1][1] = wmma_bf16(a1, b1, acc[1][1]);
    acc[1][2] = wmma_bf16(a1, b2, acc[1][2]);
    acc[1][3] = wmma_bf16(a1, b3, acc[1][3]);
  }
#pragma unroll
  for (int mi = 0; mi < 2; ++mi)
#pragma unroll
    for (int t = 0; t < 4; ++t) {
      int col = n0 + (t << 4) + (lane & 15);
      float bv = bias[col];
#pragma unroll
      for (int i = 0; i < 8; ++i) {
        int row = m0 + (mi << 4) + i + ((lane >> 4) << 3);
        out[(size_t)row * CHN + col] = acc[mi][t][i] + bv;
      }
    }
}

extern "C" void kernel_launch(void* const* d_in, const int* in_sizes, int n_in,
                              void* d_out, int out_size, void* d_ws, size_t ws_size,
                              hipStream_t stream) {
  const int HS[4]  = {64, 32, 16, 8};
  const int CH[4]  = {128, 256, 512, 1024};
  const int LHS[4] = {6, 5, 4, 3};   // log2(HS)
  const int LWS[4] = {3, 2, 1, 0};   // log2(ws)
  const int OFF[4] = {0, 64, 80, 84};

  char* ws = (char*)d_ws;
  __bf16* wt = (__bf16*)ws;                     // bf16 transposed weights
  const size_t fcWt_off[4]  = {0, 16384, 49152, 114688};
  const size_t qkvWt_off = 245760, projWt_off = 294912;
  const size_t fc1Wt_off = 311296, fc2Wt_off = 376832;
  const size_t revWt_off[4] = {442368, 458752, 491520, 557056};  // end 688128

  float*  xcat = (float*)(ws + 1376256);        // [TTOT][128] f32 (token buffer)
  __bf16* qbuf = (__bf16*)(ws + 45940736);      // [TTOT][384] qkv, later [TTOT][512] h1
  __bf16* xln  = (__bf16*)(ws + 135069696);     // [TTOT][128] bf16
  __bf16* aout = (__bf16*)(ws + 157351936);     // [TTOT][128] bf16

  // 1) convert+transpose all weights to bf16 [N][K]
  struct WTDesc { const float* W; __bf16* T; int K, N; } wtd[12] = {
    {(const float*)d_in[1],  wt + fcWt_off[0], 128,  128},
    {(const float*)d_in[6],  wt + fcWt_off[1], 256,  128},
    {(const float*)d_in[11], wt + fcWt_off[2], 512,  128},
    {(const float*)d_in[16], wt + fcWt_off[3], 1024, 128},
    {(const float*)d_in[3],  wt + revWt_off[0], 128, 128},
    {(const float*)d_in[8],  wt + revWt_off[1], 128, 256},
    {(const float*)d_in[13], wt + revWt_off[2], 128, 512},
    {(const float*)d_in[18], wt + revWt_off[3], 128, 1024},
    {(const float*)d_in[24], wt + qkvWt_off,   128, 384},
    {(const float*)d_in[25], wt + projWt_off,  128, 128},
    {(const float*)d_in[26], wt + fc1Wt_off,   128, 512},
    {(const float*)d_in[28], wt + fc2Wt_off,   512, 128},
  };
  for (int i = 0; i < 12; ++i) {
    int tot = wtd[i].K * wtd[i].N;
    wtrans_kernel<<<(tot + 255) / 256, 256, 0, stream>>>(wtd[i].W, wtd[i].T,
                                                         wtd[i].K, wtd[i].N);
  }

  // 2) per-scale fc + window partition into xcat
  for (int j = 0; j < 4; ++j) {
    int M = 16 * HS[j] * HS[j];
    fc_gemm<<<dim3(M / 128, 2), 128, 0, stream>>>(
        (const float*)d_in[5 * j], wt + fcWt_off[j], (const float*)d_in[5 * j + 2],
        xcat, CH[j], LHS[j], LWS[j], OFF[j]);
  }
  // 3) LN1
  ln_kernel<<<TTOT / 8, 256, 0, stream>>>(xcat, (const float*)d_in[20],
                                          (const float*)d_in[21], xln);
  // 4) QKV projection (no bias)
  gemm_bf16<<<dim3(TTOT / 128, 6), 128, 0, stream>>>(xln, wt + qkvWt_off, nullptr,
                                                     qbuf, 384, 128, 0);
  // 5) windowed multi-head attention
  attn_kernel<<<NWIN, 256, 0, stream>>>(qbuf, aout);
  // 6) proj + residual (in place into xcat)
  gemm_f32_resid<<<dim3(TTOT / 128, 2), 128, 0, stream>>>(aout, wt + projWt_off,
                                                          nullptr, xcat, 128, 128);
  // 7) LN2
  ln_kernel<<<TTOT / 8, 256, 0, stream>>>(xcat, (const float*)d_in[22],
                                          (const float*)d_in[23], xln);
  // 8) MLP fc1 + relu
  gemm_bf16<<<dim3(TTOT / 128, 8), 128, 0, stream>>>(xln, wt + fc1Wt_off,
                                                     (const float*)d_in[27], qbuf,
                                                     512, 128, 1);
  // 9) MLP fc2 + residual
  gemm_f32_resid<<<dim3(TTOT / 128, 2), 128, 0, stream>>>(qbuf, wt + fc2Wt_off,
                                                          (const float*)d_in[29],
                                                          xcat, 128, 512);
  // 10) per-scale window-reverse + rev linear -> d_out (concatenated)
  float* outp = (float*)d_out;
  size_t ooff = 0;
  for (int j = 0; j < 4; ++j) {
    int M = 16 * HS[j] * HS[j];
    rev_gemm<<<dim3(M / 128, CH[j] / 64), 128, 0, stream>>>(
        xcat, wt + revWt_off[j], (const float*)d_in[5 * j + 4], outp + ooff,
        CH[j], LHS[j], LWS[j], OFF[j]);
    ooff += (size_t)M * CH[j];
  }
}
